// Spiking_NeuLF_snn_61924838473792
// MI455X (gfx1250) — matrix operand order repaired and checked by reading, not compiled, not run on hardware
//
#include <hip/hip_runtime.h>

typedef __attribute__((ext_vector_type(8)))  float  v8f;
typedef __attribute__((ext_vector_type(8)))  __bf16 v8bf;
typedef __attribute__((ext_vector_type(16))) __bf16 v16bf;

#define T_STEPS 250
#define N_ROWS  4096
#define SW      264   // LDS row stride (bf16 elems); 264*2=528 bytes, 16B aligned

// ---- d_ws layout ----
// [0 .. MEM_FLOATS*4)            : f32 membrane state (zeroed each launch)
//   mem_in  : [4096][256] row-major
//   memp[i] : 4096*256 each, fragment-major (tile(mtg,nt) of 256 f32, idx lane*8+r)
//   mem_v   : 4096*128 fragment-major
// [+0 .. +1376256)               : bf16 weights, fragment-major
// [+1376256 .. +1384448)         : packed hidden-layer biases, 8*256 f32
#define MEM_FLOATS   ((size_t)N_ROWS * 256 * 9 + (size_t)N_ROWS * 128)
#define WB_ELEMS     ((size_t)688128)
#define WFEAT_OFF    ((size_t)589824)
#define WVIEW_OFF    ((size_t)655360)

struct Biases { const float* p[8]; };
union ABf { v16bf v; v8bf h[2]; };

// Opaque 64-bit value: defeats LICM (forces re-materialization each loop
// iteration) without destroying pointer address-space provenance.
__device__ __forceinline__ size_t opaque(size_t v) {
  asm volatile("" : "+s"(v));
  return v;
}

// ------------------------------------------------------------------
// Zero membrane state + output accumulator (graph-safe, every launch)
// ------------------------------------------------------------------
__global__ void zero_kernel(float* __restrict__ mem, unsigned long long n,
                            float* __restrict__ out) {
  unsigned long long i = (unsigned long long)blockIdx.x * blockDim.x + threadIdx.x;
  if (i == 0) out[0] = 0.0f;
  unsigned long long stride = (unsigned long long)gridDim.x * blockDim.x;
  for (; i < n; i += stride) mem[i] = 0.0f;
}

// Pack the 8 hidden-layer bias vectors contiguously (avoids dynamic
// kernarg-struct indexing / scratch in the main kernel's rolled loop).
__global__ void pack_biases(Biases bp, float* __restrict__ dst) {
  int tid = blockIdx.x * blockDim.x + threadIdx.x;  // 0..2047
  dst[tid] = bp.p[tid >> 8][tid & 255];
}

// ------------------------------------------------------------------
// Convert a [Nd x Kd] f32 weight (row-major, W[n][k]) into bf16
// B-fragment-major layout: fragment f = kt*nn + nt holds 512 bf16;
// lane l, elem i: B(K = kt*32 + (l>>4)*16 + i, N = nt*16 + (l&15))
// ------------------------------------------------------------------
__global__ void conv_weights(const float* __restrict__ src, __bf16* __restrict__ dst,
                             int Kd, int Nd) {
  int tid = blockIdx.x * blockDim.x + threadIdx.x;
  int nn = Nd >> 4, nk = Kd >> 5;
  int f = tid >> 5, l = tid & 31;
  if (f >= nk * nn) return;
  int kt = f / nn, nt = f % nn;
  const float* s = src + (size_t)(nt * 16 + (l & 15)) * Kd + kt * 32 + (l >> 4) * 16;
  __bf16* d = dst + (size_t)f * 512 + l * 16;
#pragma unroll
  for (int i = 0; i < 16; ++i) d[i] = (__bf16)s[i];
}

// ------------------------------------------------------------------
// One GEMM layer on a 32-row tile. MODE: 0 = LIF + spike store (bf16 LDS),
// 1 = feature (bias only, bf16 LDS), 2 = view (LIF + spike-count accumulate).
// NTW = N-tiles per wave (4 -> 256 cols, 2 -> 128 cols).
// ------------------------------------------------------------------
template <int MODE, int NTW>
__device__ __forceinline__ void layer_gemm(
    const __bf16* __restrict__ aLo,   // LDS, K 0..255
    const __bf16* __restrict__ aHi,   // LDS, K 256..511 (skip concat)
    const __bf16* __restrict__ wfrag, // fragment-major bf16 weights (global/L2)
    const float*  __restrict__ bias,
    float*        __restrict__ mem,   // fragment-major f32 state (MODE 0/2)
    __bf16*       __restrict__ outBuf,
    int nk, int rowTile0, float& spkAcc) {
  const int tid  = threadIdx.x;
  const int wave = tid >> 5, lane = tid & 31;
  const int mt   = wave >> 2;
  const int lm   = lane & 15, lg = lane >> 4;
  const int nn   = NTW * 4;

  v8f c[NTW];
#pragma unroll
  for (int j = 0; j < NTW; ++j) c[j] = (v8f){0,0,0,0,0,0,0,0};

  const int arow = mt * 16 + lm;
#pragma unroll 1
  for (int kt = 0; kt < nk; ++kt) {
    const __bf16* ab = (kt < 8) ? aLo : aHi;
    const int kb = (kt & 7) * 32 + lg * 16;
    ABf a;
    a.h[0] = *(const v8bf*)(ab + arow * SW + kb);
    a.h[1] = *(const v8bf*)(ab + arow * SW + kb + 8);
#pragma unroll
    for (int j = 0; j < NTW; ++j) {
      const int nt = (wave & 3) + 4 * j;
      v16bf b = *(const v16bf*)(wfrag + ((size_t)(kt * nn + nt) * 512) + lane * 16);
      c[j] = __builtin_amdgcn_wmma_f32_16x16x32_bf16(
          false, a.v, false, b, (short)0, c[j], false, false);
    }
  }

#pragma unroll
  for (int j = 0; j < NTW; ++j) {
    const int nt   = (wave & 3) + 4 * j;
    const int chan = nt * 16 + lm;
    const float bsv = bias[chan];
    if constexpr (MODE == 1) {
#pragma unroll
      for (int r = 0; r < 8; ++r) {
        int row = mt * 16 + lg * 8 + r;
        outBuf[row * SW + chan] = (__bf16)(c[j][r] + bsv);
      }
    } else {
      float* mp = mem + (((size_t)(rowTile0 + mt) * nn + nt) * 256) + lane * 8;
#pragma unroll
      for (int r = 0; r < 8; ++r) {
        float m     = mp[r];
        float reset = (m > 1.0f) ? 1.0f : 0.0f;           // spike(mem - THR), detached
        float mn    = 0.9f * m + (c[j][r] + bsv) - reset; // beta*mem + inp - reset*thr
        mp[r] = mn;
        float s = (mn > 1.0f) ? 1.0f : 0.0f;
        if constexpr (MODE == 0) {
          int row = mt * 16 + lg * 8 + r;
          outBuf[row * SW + chan] = (__bf16)s;
        } else {
          spkAcc += s;
        }
      }
    }
  }
}

// ------------------------------------------------------------------
// Persistent per-row-tile SNN kernel: 128 blocks x 256 threads,
// each block owns rows [blockIdx*32, +32) for all 250 steps.
// ------------------------------------------------------------------
__global__ __launch_bounds__(256) void snn_main(
    const float* __restrict__ x, const float* __restrict__ W_in,
    const float* __restrict__ b_in, const float* __restrict__ biasAll,
    const float* __restrict__ b_feat, const float* __restrict__ b_view,
    const __bf16* __restrict__ wb, float* __restrict__ mem,
    float* __restrict__ out) {
  __shared__ __align__(16) __bf16 sIn[32 * SW];
  __shared__ __align__(16) __bf16 sA[32 * SW];
  __shared__ __align__(16) __bf16 sB[32 * SW];
  __shared__ float sred[256];

  const int rowBase  = blockIdx.x * 32;
  const int rowTile0 = blockIdx.x * 2;
  float* mem_in = mem;                                 // row-major
  float* mem_v  = mem + (size_t)N_ROWS * 256 * 9;      // fragment-major
  float spkAcc = 0.0f, dummy = 0.0f;

  const int tid = threadIdx.x;
  const int ir  = tid >> 3;            // 0..31 local row
  const int ic0 = (tid & 7) * 32;      // 32-channel slab

#pragma unroll 1
  for (int t = 0; t < T_STEPS; ++t) {
    // --- input layer: inp = xt @ W_in.T + b_in ; LIF(mem_in) -> spikes in sIn ---
    {
      const int row = rowBase + ir;
      const float4 xv = *(const float4*)(x + ((size_t)t * N_ROWS + row) * 4);
      float* mrow = mem_in + (size_t)row * 256;
#pragma unroll 4
      for (int c = 0; c < 32; ++c) {
        int chan = ic0 + c;
        const float* wr = W_in + chan * 4;
        float inp = b_in[chan] + xv.x * wr[0] + xv.y * wr[1] + xv.z * wr[2] + xv.w * wr[3];
        float m     = mrow[chan];
        float reset = (m > 1.0f) ? 1.0f : 0.0f;
        float mn    = 0.9f * m + inp - reset;
        mrow[chan]  = mn;
        sIn[ir * SW + chan] = (__bf16)((mn > 1.0f) ? 1.0f : 0.0f);
      }
    }
    __syncthreads();

    // --- 8 hidden layers (layer 5 takes 512-wide concat [h, input_pts]) ---
    const __bf16* src = sIn;
    __bf16* dst = sA;
#pragma unroll 1
    for (int i = 0; i < 8; ++i) {
      // opaque(): loop-variant weight offset -> loads stream from L2 each
      // step instead of being hoisted across t and spilled to scratch.
      const size_t woff =
          opaque((size_t)i * 65536 + (i > 5 ? (size_t)65536 : 0));
      const int nk = (i == 5) ? 16 : 8;
      float* memp = mem + (size_t)N_ROWS * 256 * (1 + i);
      layer_gemm<0, 4>(src, sIn, wb + woff, biasAll + i * 256, memp, dst, nk,
                       rowTile0, dummy);
      __syncthreads();
      src = dst;
      dst = (dst == sA) ? sB : sA;
    }

    // --- feature (no LIF), bf16 into free buffer ---
    layer_gemm<1, 4>(src, src, wb + opaque(WFEAT_OFF), b_feat, nullptr, dst, 8,
                     rowTile0, dummy);
    __syncthreads();

    // --- view: 128 cols, LIF(mem_v), accumulate spike count ---
    layer_gemm<2, 2>(dst, dst, wb + opaque(WVIEW_OFF), b_view, mem_v, nullptr, 8,
                     rowTile0, spkAcc);
    __syncthreads();
  }

  // block reduction + scale by 1/T
  sred[tid] = spkAcc;
  __syncthreads();
#pragma unroll
  for (int s = 128; s > 0; s >>= 1) {
    if (tid < s) sred[tid] += sred[tid + s];
    __syncthreads();
  }
  if (tid == 0) atomicAdd(out, sred[0] * (1.0f / (float)T_STEPS));
}

// ------------------------------------------------------------------
extern "C" void kernel_launch(void* const* d_in, const int* in_sizes, int n_in,
                              void* d_out, int out_size, void* d_ws, size_t ws_size,
                              hipStream_t stream) {
  // dict order: x, W_in, b_in, (Wp_i, bp_i)*8, W_feat, b_feat, W_view, b_view
  const float* x     = (const float*)d_in[0];
  const float* W_in  = (const float*)d_in[1];
  const float* b_in  = (const float*)d_in[2];
  const float* bfeat = (const float*)d_in[20];
  const float* bview = (const float*)d_in[22];
  Biases bp;
  for (int i = 0; i < 8; ++i) bp.p[i] = (const float*)d_in[4 + 2 * i];

  float*  mem     = (float*)d_ws;
  __bf16* wb      = (__bf16*)((char*)d_ws + MEM_FLOATS * 4);
  float*  biasAll = (float*)((char*)d_ws + MEM_FLOATS * 4 + WB_ELEMS * 2);

  zero_kernel<<<2048, 256, 0, stream>>>(mem, (unsigned long long)MEM_FLOATS, (float*)d_out);
  pack_biases<<<8, 256, 0, stream>>>(bp, biasAll);

  // weight conversions: {d_in idx, bf16 offset, Kd, Nd}
  const struct { int src; size_t off; int Kd, Nd; } cv[10] = {
      {3, 0, 256, 256},      {5, 65536, 256, 256},  {7, 131072, 256, 256},
      {9, 196608, 256, 256}, {11, 262144, 256, 256},{13, 327680, 512, 256},
      {15, 458752, 256, 256},{17, 524288, 256, 256},
      {19, WFEAT_OFF, 256, 256}, {21, WVIEW_OFF, 256, 128}};
  for (int i = 0; i < 10; ++i) {
    int nfrag   = (cv[i].Kd >> 5) * (cv[i].Nd >> 4);
    int threads = nfrag * 32;
    conv_weights<<<(threads + 255) / 256, 256, 0, stream>>>(
        (const float*)d_in[cv[i].src], wb + cv[i].off, cv[i].Kd, cv[i].Nd);
  }

  snn_main<<<N_ROWS / 32, 256, 0, stream>>>(x, W_in, b_in, biasAll, bfeat, bview, wb,
                                            mem, (float*)d_out);
}
